// AVWGCN_14972255993886
// MI455X (gfx1250) — compile-verified, matrix operand order checked by reading, not run on hardware
//
#include <hip/hip_runtime.h>

typedef __attribute__((ext_vector_type(16))) _Float16 v16h;
typedef __attribute__((ext_vector_type(8)))  _Float16 v8h;
typedef __attribute__((ext_vector_type(8)))  float    v8f;

#define CAT16(lo, hi) __builtin_shufflevector((lo), (hi), 0,1,2,3,4,5,6,7,8,9,10,11,12,13,14,15)

// CDNA5 async global->LDS copy (16B per lane), tracked by ASYNCcnt.
__device__ __forceinline__ void async_copy_b128(const void* gptr, void* lptr)
{
    const unsigned int la = (unsigned int)(uintptr_t)lptr;  // low 32 bits = LDS offset
    asm volatile("global_load_async_to_lds_b128 %0, %1, off"
                 :: "v"(la), "v"(gptr) : "memory");
}
__device__ __forceinline__ void wait_async()
{
    asm volatile("s_wait_asynccnt 0" ::: "memory");
}

// Problem constants: B=64, N=2048, CI=32, CO=64, K=3, D=16, RED=16

// ---------------------------------------------------------------------------
// NT WMMA GEMM: C = A(f16, [M x Kd] row-major) * Bt^T (Bt: f16, [Nc x Kd] row-major)
// 128x64 block tile, 256 threads = 8 waves (4 M-waves x 2 N-waves).
// Each wave: 2x2 grid of 16x16 accumulators -> 4 WMMAs per 32-deep k-step.
// Double-buffered LDS with async global->LDS staging; 1 barrier/step.
// Fused epilogues (compile-time EMODE):
//  0: T2 = 2*C - I                    -> f16 [M x Nc]
//  1: supports@X -> Xg f16 [n][b][KSLOT*32+c]      (col = b*32+c)
//  2: cheb@X1    -> Xg1 f16 [b][n][KSLOT*64+o]     (col = b*64+o)
//  3: x@initW+b  -> Xg1 slot0 f16 AND X1col f16 [node][b*64+o]
//  4: Xg1@gconvW+b, leakyReLU -> f32 [M x 64]
// ---------------------------------------------------------------------------
template <int EMODE, int KSLOT>
__global__ __launch_bounds__(256) void wmma_gemm_nt_kernel(
    const _Float16* __restrict__ A, int lda,
    const _Float16* __restrict__ Bt, int ldb,
    int Nc, int Kd,
    const float* __restrict__ bias,
    _Float16* __restrict__ outH,
    _Float16* __restrict__ outH2,
    float* __restrict__ outF)
{
    __shared__ alignas(16) _Float16 Asl[2][128 * 32];  // 8KB x2
    __shared__ alignas(16) _Float16 Bsl[2][64 * 32];   // 4KB x2

    const int tid  = threadIdx.x;
    const int lane = tid & 31;
    const int wid  = tid >> 5;
    const int wm   = wid & 3;   // M-wave: rows [wm*32, wm*32+32)
    const int wn   = wid >> 2;  // N-wave: cols [wn*32, wn*32+32)
    const int hf   = lane >> 4;
    const int lr   = lane & 15;
    const int kb   = hf * 8;

    const int tile_m0 = blockIdx.x * 128;
    const int tile_n0 = blockIdx.y * 64;

    auto stage = [&](int buf, int k0) {
#pragma unroll
        for (int i = 0; i < 2; ++i) {                 // A: 512 16B chunks
            int ck  = tid + i * 256;
            int row = ck >> 2, ch = (ck & 3) * 8;
            async_copy_b128(A + (size_t)(tile_m0 + row) * lda + k0 + ch,
                            &Asl[buf][row * 32 + ch]);
        }
        {                                             // B: 256 16B chunks
            int row = tid >> 2, ch = (tid & 3) * 8;
            async_copy_b128(Bt + (size_t)(tile_n0 + row) * ldb + k0 + ch,
                            &Bsl[buf][row * 32 + ch]);
        }
    };

    // 16-bit A-layout fragment: lanes 0-15 hold K {0-7,16-23}, lanes 16-31 K {8-15,24-31}
    auto frag = [&](const _Float16* S, int m) -> v16h {
        v8h lo = *(const v8h*)(&S[m * 32 + kb]);
        v8h hi = *(const v8h*)(&S[m * 32 + 16 + kb]);
        return CAT16(lo, hi);
    };

    v8f acc00 = {}, acc01 = {}, acc10 = {}, acc11 = {};

    stage(0, 0);
    wait_async();
    __syncthreads();

    const int nsteps = Kd >> 5;
    for (int s = 0; s < nsteps; ++s) {
        const int buf = s & 1;
        if (s + 1 < nsteps) stage(buf ^ 1, (s + 1) << 5);
        if (s + 2 < nsteps) {
            const int k2 = (s + 2) << 5;
            __builtin_prefetch(A + (size_t)(tile_m0 + (tid >> 2)) * lda + k2, 0, 3);
            __builtin_prefetch(Bt + (size_t)(tile_n0 + (tid >> 2)) * ldb + k2, 0, 3);
        }

        v16h a0 = frag(Asl[buf], wm * 32 + lr);
        v16h a1 = frag(Asl[buf], wm * 32 + 16 + lr);
        v16h b0 = frag(Bsl[buf], wn * 32 + lr);
        v16h b1 = frag(Bsl[buf], wn * 32 + 16 + lr);

        acc00 = __builtin_amdgcn_wmma_f32_16x16x32_f16(false, a0, false, b0, (short)0, acc00, false, false);
        acc01 = __builtin_amdgcn_wmma_f32_16x16x32_f16(false, a0, false, b1, (short)0, acc01, false, false);
        acc10 = __builtin_amdgcn_wmma_f32_16x16x32_f16(false, a1, false, b0, (short)0, acc10, false, false);
        acc11 = __builtin_amdgcn_wmma_f32_16x16x32_f16(false, a1, false, b1, (short)0, acc11, false, false);

        wait_async();
        __syncthreads();
    }

    // epilogue; C layout: vgpr r, lane l -> M = r + (l<16?0:8), N = l%16
#pragma unroll
    for (int mm = 0; mm < 2; ++mm) {
#pragma unroll
        for (int nn = 0; nn < 2; ++nn) {
            v8f acc = (mm == 0) ? (nn == 0 ? acc00 : acc01)
                                : (nn == 0 ? acc10 : acc11);
            const int gn = tile_n0 + wn * 32 + nn * 16 + lr;
            float bv = 0.0f;
            if constexpr (EMODE == 3 || EMODE == 4) bv = bias[gn];
#pragma unroll
            for (int r = 0; r < 8; ++r) {
                const int gm = tile_m0 + wm * 32 + mm * 16 + hf * 8 + r;
                float v = acc[r] + bv;
                if constexpr (EMODE == 0) {
                    float t = 2.0f * v - ((gm == gn) ? 1.0f : 0.0f);
                    outH[(size_t)gm * Nc + gn] = (_Float16)t;
                } else if constexpr (EMODE == 1) {
                    int bb = gn >> 5, c = gn & 31;
                    outH[((size_t)gm * 64 + bb) * 96 + KSLOT * 32 + c] = (_Float16)v;
                } else if constexpr (EMODE == 2) {
                    int bb = gn >> 6, o = gn & 63;
                    outH[((size_t)bb * 2048 + gm) * 192 + KSLOT * 64 + o] = (_Float16)v;
                } else if constexpr (EMODE == 3) {
                    int node = gm & 2047, bb = gm >> 11;
                    _Float16 hv = (_Float16)v;
                    outH[(size_t)gm * 192 + gn]               = hv;  // Xg1 slot k=0
                    outH2[(size_t)node * 4096 + bb * 64 + gn] = hv;  // X1col
                } else {  // EMODE == 4
                    float t = (v >= 0.0f) ? v : 0.01f * v;
                    outF[(size_t)gm * 64 + gn] = t;
                }
            }
        }
    }
}

// ---------------------------------------------------------------------------
// Per-node adaptive gconv via hypernetwork weights built in LDS (never hit HBM).
// A-slab copied with async DMA, overlapped with the hypernetwork weight build.
// 512 threads = 16 waves; 4x4 grid of 16x16 tiles; Kd = 96 (3 WMMA steps).
// ---------------------------------------------------------------------------
__global__ __launch_bounds__(512) void node_gconv_kernel(
    const _Float16* __restrict__ Xg,    // [N][64][96] f16
    const float* __restrict__ emb,      // [N][16]
    const float* __restrict__ wpool,    // [16][3*32*64]
    const float* __restrict__ bpool,    // [16][64]
    float* __restrict__ lrelu0)         // [B][N][64] f32
{
    __shared__ alignas(16) _Float16 Al[64 * 96];
    __shared__ alignas(16) _Float16 Wt[64 * 96];   // transposed [o][kci]
    __shared__ float bnode[64];
    __shared__ float e[16];

    const int n   = blockIdx.x;
    const int tid = threadIdx.x;

    // issue async A-slab copy first (768 16B chunks), overlap with weight build
    {
        const _Float16* src = Xg + (size_t)n * 64 * 96;
        for (int i = tid; i < 64 * 96 / 8; i += 512)
            async_copy_b128(src + i * 8, &Al[i * 8]);
    }

    if (tid < 16) e[tid] = emb[n * 16 + tid];
    __syncthreads();

    for (int i = tid; i < 6144; i += 512) {
        int kci = i >> 6, o = i & 63;
        float s = 0.0f;
#pragma unroll
        for (int d = 0; d < 16; ++d) s += e[d] * wpool[d * 6144 + kci * 64 + o];
        Wt[o * 96 + kci] = (_Float16)s;
    }
    if (tid < 64) {
        float s = 0.0f;
#pragma unroll
        for (int d = 0; d < 16; ++d) s += e[d] * bpool[d * 64 + tid];
        bnode[tid] = s;
    }
    wait_async();
    __syncthreads();

    const int lane = tid & 31;
    const int wid  = tid >> 5;
    const int wm   = wid & 3;
    const int wn   = wid >> 2;  // 0..3
    const int hf   = lane >> 4;
    const int lr   = lane & 15;
    const int kb   = hf * 8;

    v8f acc = {};
#pragma unroll
    for (int ks = 0; ks < 3; ++ks) {
        const int am = wm * 16 + lr;
        v8h alo = *(const v8h*)(&Al[am * 96 + ks * 32 + kb]);
        v8h ahi = *(const v8h*)(&Al[am * 96 + ks * 32 + 16 + kb]);
        v16h af = CAT16(alo, ahi);

        const int bn = wn * 16 + lr;
        v8h blo = *(const v8h*)(&Wt[bn * 96 + ks * 32 + kb]);
        v8h bhi = *(const v8h*)(&Wt[bn * 96 + ks * 32 + 16 + kb]);
        v16h bf = CAT16(blo, bhi);

        acc = __builtin_amdgcn_wmma_f32_16x16x32_f16(false, af, false, bf, (short)0, acc, false, false);
    }

#pragma unroll
    for (int r = 0; r < 8; ++r) {
        const int b = wm * 16 + hf * 8 + r;
        const int o = wn * 16 + lr;
        float v = acc[r] + bnode[o];
        v = (v >= 0.0f) ? v : 0.01f * v;
        lrelu0[((size_t)b * 2048 + n) * 64 + o] = v;
    }
}

// ---------------------------------------------------------------------------
// Staging helpers
// ---------------------------------------------------------------------------
__global__ void cvt_f16_kernel(const float* __restrict__ src, _Float16* __restrict__ dst, int count)
{
    for (int i = blockIdx.x * blockDim.x + threadIdx.x; i < count; i += gridDim.x * blockDim.x)
        dst[i] = (_Float16)src[i];
}

// Batched LDS-tiled transpose, f32 [z][R][C] -> f16 [z][C][R]; R,C multiples of 32.
__global__ __launch_bounds__(256) void transpose_f32_f16_kernel(
    const float* __restrict__ src, _Float16* __restrict__ dst, int R, int C)
{
    __shared__ _Float16 t[32][33];
    const int c0 = blockIdx.x * 32;
    const int r0 = blockIdx.y * 32;
    const size_t base = (size_t)blockIdx.z * R * C;
    const int tx = threadIdx.x & 31;
    const int ty = threadIdx.x >> 5;   // 0..7
#pragma unroll
    for (int yy = ty; yy < 32; yy += 8)
        t[yy][tx] = (_Float16)src[base + (size_t)(r0 + yy) * C + c0 + tx];
    __syncthreads();
#pragma unroll
    for (int yy = ty; yy < 32; yy += 8)
        dst[base + (size_t)(c0 + yy) * R + r0 + tx] = t[tx][yy];
}

// LDS-tiled transpose, f16 [R][C] -> f16 [C][R]; R,C multiples of 32.
__global__ __launch_bounds__(256) void transpose_f16_kernel(
    const _Float16* __restrict__ src, _Float16* __restrict__ dst, int R, int C)
{
    __shared__ _Float16 t[32][33];
    const int c0 = blockIdx.x * 32;
    const int r0 = blockIdx.y * 32;
    const int tx = threadIdx.x & 31;
    const int ty = threadIdx.x >> 5;
#pragma unroll
    for (int yy = ty; yy < 32; yy += 8)
        t[yy][tx] = src[(size_t)(r0 + yy) * C + c0 + tx];
    __syncthreads();
#pragma unroll
    for (int yy = ty; yy < 32; yy += 8)
        dst[(size_t)(c0 + yy) * R + r0 + tx] = t[tx][yy];
}

// x[b,n,c] -> Xg k=0 slot [n][b][0..31]
__global__ void scatter_x_kernel(const float* __restrict__ x, _Float16* __restrict__ Xg)
{
    const int total = 64 * 2048 * 32;
    for (int idx = blockIdx.x * blockDim.x + threadIdx.x; idx < total; idx += gridDim.x * blockDim.x) {
        int b = idx >> 16;            // 2048*32 = 65536
        int rem = idx & 65535;
        int n = rem >> 5, c = rem & 31;
        Xg[((size_t)n * 64 + b) * 96 + c] = (_Float16)x[idx];
    }
}

// y[b*64+co] = mean_n src[(b*2048+n)*64+co]
__global__ __launch_bounds__(256) void colmean_kernel(const float* __restrict__ src, float* __restrict__ y)
{
    const int b = blockIdx.x >> 6, co = blockIdx.x & 63;
    const float* p = src + (size_t)b * 2048 * 64 + co;
    float s = 0.0f;
    for (int n = threadIdx.x; n < 2048; n += 256) s += p[(size_t)n * 64];
    __shared__ float red[256];
    red[threadIdx.x] = s;
    __syncthreads();
    for (int st = 128; st > 0; st >>= 1) {
        if (threadIdx.x < st) red[threadIdx.x] += red[threadIdx.x + st];
        __syncthreads();
    }
    if (threadIdx.x == 0) y[blockIdx.x] = red[0] * (1.0f / 2048.0f);
}

// SelfAtt MLP head -> hardsigmoid scalar per batch, both branches
__global__ void att_kernel(const float* __restrict__ y0, const float* __restrict__ y1,
                           const float* __restrict__ f11, const float* __restrict__ f12,
                           const float* __restrict__ f21, const float* __restrict__ f22,
                           float* __restrict__ s1, float* __restrict__ s2)
{
    const int b = threadIdx.x;  // blockDim = 64
    if (b >= 64) return;
    {
        float acc = 0.0f;
        for (int j = 0; j < 4; ++j) {
            float h = 0.0f;
            for (int o = 0; o < 64; ++o) h += y0[b * 64 + o] * f11[o * 4 + j];
            acc += fmaxf(h, 0.0f) * f12[j];
        }
        s1[b] = fminf(fmaxf(acc * (1.0f / 6.0f) + 0.5f, 0.0f), 1.0f);
    }
    {
        float acc = 0.0f;
        for (int j = 0; j < 4; ++j) {
            float h = 0.0f;
            for (int o = 0; o < 64; ++o) h += y1[b * 64 + o] * f21[o * 4 + j];
            acc += fmaxf(h, 0.0f) * f22[j];
        }
        s2[b] = fminf(fmaxf(acc * (1.0f / 6.0f) + 0.5f, 0.0f), 1.0f);
    }
}

__global__ void combine_kernel(const float* __restrict__ l0, const float* __restrict__ l1,
                               const float* __restrict__ s1, const float* __restrict__ s2,
                               float* __restrict__ out, int total)
{
    for (int idx = blockIdx.x * blockDim.x + threadIdx.x; idx < total; idx += gridDim.x * blockDim.x) {
        int b = idx >> 17;  // 2048*64 = 131072
        out[idx] = l0[idx] * s1[b] + l1[idx] * s2[b];
    }
}

// ---------------------------------------------------------------------------
extern "C" void kernel_launch(void* const* d_in, const int* in_sizes, int n_in,
                              void* d_out, int out_size, void* d_ws, size_t ws_size,
                              hipStream_t stream)
{
    (void)in_sizes; (void)n_in; (void)out_size; (void)ws_size;

    const float* x       = (const float*)d_in[0];
    const float* emb     = (const float*)d_in[1];
    const float* Lt      = (const float*)d_in[2];
    const float* cheb    = (const float*)d_in[3];
    const float* wpool   = (const float*)d_in[4];
    const float* bpool   = (const float*)d_in[5];
    const float* init_w  = (const float*)d_in[6];
    const float* init_b  = (const float*)d_in[7];
    const float* gconv_w = (const float*)d_in[8];
    const float* gconv_b = (const float*)d_in[9];
    const float* g1f1    = (const float*)d_in[10];
    const float* g1f2    = (const float*)d_in[11];
    const float* g2f1    = (const float*)d_in[12];
    const float* g2f2    = (const float*)d_in[13];
    float* out = (float*)d_out;

    char* ws = (char*)d_ws;
    size_t off = 0;
    auto take = [&](size_t bytes) -> char* {
        char* p = ws + off;
        off += (bytes + 255) & ~(size_t)255;
        return p;
    };

    const size_t NN = 2048ull * 2048ull;
    _Float16* Lf16   = (_Float16*)take(NN * 2);                // L (A operand)
    _Float16* LT16   = (_Float16*)take(NN * 2);                // L^T (B operand)
    _Float16* T2f16  = (_Float16*)take(NN * 2);                // 2L^2 - I
    _Float16* C1f16  = (_Float16*)take(NN * 2);                // cheb_polys[1]
    _Float16* C2f16  = (_Float16*)take(NN * 2);                // cheb_polys[2]
    _Float16* Xrow   = (_Float16*)take(131072ull * 32 * 2);    // x as [B*N, 32]
    _Float16* XcolT  = (_Float16*)take(NN * 2);                // [b*32+c][n]
    _Float16* Xg     = (_Float16*)take(2048ull * 64 * 96 * 2); // [n][b][k*32+c]
    _Float16* X1col  = (_Float16*)take(2048ull * 4096 * 2);    // [n][b*64+o]
    _Float16* X1colT = (_Float16*)take(2048ull * 4096 * 2);    // [b*64+o][n]
    _Float16* Xg1    = (_Float16*)take(131072ull * 192 * 2);   // [b*N+n][k*64+o]
    _Float16* iWT    = (_Float16*)take(64ull * 32 * 2);        // init_w^T
    _Float16* gWT    = (_Float16*)take(64ull * 192 * 2);       // gconv_w^T
    float*    lrelu0 = (float*)take(131072ull * 64 * 4);
    float*    lrelu1 = (float*)take(131072ull * 64 * 4);
    float*    y0     = (float*)take(64ull * 64 * 4);
    float*    y1     = (float*)take(64ull * 64 * 4);
    float*    s1     = (float*)take(64ull * 4);
    float*    s2     = (float*)take(64ull * 4);

    // ---- stage f16 operands (A row-major; all B operands pre-transposed) ----
    cvt_f16_kernel<<<4096, 256, 0, stream>>>(Lt, Lf16, (int)NN);
    cvt_f16_kernel<<<4096, 256, 0, stream>>>(cheb + NN, C1f16, (int)NN);
    cvt_f16_kernel<<<4096, 256, 0, stream>>>(cheb + 2 * NN, C2f16, (int)NN);
    cvt_f16_kernel<<<4096, 256, 0, stream>>>(x, Xrow, 131072 * 32);
    transpose_f32_f16_kernel<<<dim3(64, 64, 1),  256, 0, stream>>>(Lt, LT16, 2048, 2048);
    transpose_f32_f16_kernel<<<dim3(1, 64, 64),  256, 0, stream>>>(x, XcolT, 2048, 32);
    transpose_f32_f16_kernel<<<dim3(2, 1, 1),    256, 0, stream>>>(init_w, iWT, 32, 64);
    transpose_f32_f16_kernel<<<dim3(2, 6, 1),    256, 0, stream>>>(gconv_w, gWT, 192, 64);
    scatter_x_kernel<<<4096, 256, 0, stream>>>(x, Xg);

    // ---- T2 = 2*L@L - I ----
    wmma_gemm_nt_kernel<0, 0><<<dim3(16, 32), 256, 0, stream>>>(
        Lf16, 2048, LT16, 2048, 2048, 2048, nullptr, T2f16, nullptr, nullptr);

    // ---- x_g (k=1: L, k=2: T2); k=0 copied by scatter_x ----
    wmma_gemm_nt_kernel<1, 1><<<dim3(16, 32), 256, 0, stream>>>(
        Lf16, 2048, XcolT, 2048, 2048, 2048, nullptr, Xg, nullptr, nullptr);
    wmma_gemm_nt_kernel<1, 2><<<dim3(16, 32), 256, 0, stream>>>(
        T2f16, 2048, XcolT, 2048, 2048, 2048, nullptr, Xg, nullptr, nullptr);

    // ---- x1 = x @ init_w + init_b -> Xg1 slot0 + X1col ----
    wmma_gemm_nt_kernel<3, 0><<<dim3(1024, 1), 256, 0, stream>>>(
        Xrow, 32, iWT, 32, 64, 32, init_b, Xg1, X1col, nullptr);
    transpose_f16_kernel<<<dim3(128, 64), 256, 0, stream>>>(X1col, X1colT, 2048, 4096);

    // ---- x_g1 (k=1,2 via static cheb polys); k=0 done above ----
    wmma_gemm_nt_kernel<2, 1><<<dim3(16, 64), 256, 0, stream>>>(
        C1f16, 2048, X1colT, 2048, 4096, 2048, nullptr, Xg1, nullptr, nullptr);
    wmma_gemm_nt_kernel<2, 2><<<dim3(16, 64), 256, 0, stream>>>(
        C2f16, 2048, X1colT, 2048, 4096, 2048, nullptr, Xg1, nullptr, nullptr);

    // ---- x_gconv1 = Xg1 @ gconv_w + b, leakyReLU ----
    wmma_gemm_nt_kernel<4, 0><<<dim3(1024, 1), 256, 0, stream>>>(
        Xg1, 192, gWT, 192, 64, 192, gconv_b, nullptr, nullptr, lrelu1);

    // ---- x_gconv0 via per-node hypernetwork WMMA ----
    node_gconv_kernel<<<2048, 512, 0, stream>>>(Xg, emb, wpool, bpool, lrelu0);

    // ---- SelfAtt branches + combine ----
    colmean_kernel<<<4096, 256, 0, stream>>>(lrelu0, y0);
    colmean_kernel<<<4096, 256, 0, stream>>>(lrelu1, y1);
    att_kernel<<<1, 64, 0, stream>>>(y0, y1, g1f1, g1f2, g2f1, g2f2, s1, s2);
    combine_kernel<<<4096, 256, 0, stream>>>(lrelu0, lrelu1, s1, s2, out, 131072 * 64);
}
